// AttentionBias_61211873902997
// MI455X (gfx1250) — compile-verified, matrix operand order checked by reading, not run on hardware
//
#include <hip/hip_runtime.h>
#include <hip/hip_bf16.h>

typedef __attribute__((ext_vector_type(16))) _Float16 v16h;
typedef __attribute__((ext_vector_type(8)))  float    v8f;
typedef __attribute__((ext_vector_type(2)))  float    v2f;

#define NUM_HEADS 16
#define HID_VEC   64
#define NUM_GAUSS 251
#define KDIM      256   // 251 gaussians + 5 edge-type one-hot
#define WAVES_PER_BLOCK 8
#define TILE_E    16    // edges per wave

// ---------------- workspace layout (bytes) ----------------
// [0,      8192)  : Bh   : f16 WMMA-B for w_sca[:,64:320], [8 kc][32 lane][16 t]
// [8192,   8256)  : Bnorm: 16 f32
// [8256,   8320)  : C2   : 16 f32
// [8320,   9344)  : gateB: 256 f32, [4 q][32 lane][2 j]  (f32 WMMA-B of w_gate^T)

__global__ __launch_bounds__(256)
void precompute_kernel(const float* __restrict__ w_edge,   // [64]
                       const float* __restrict__ w_vec1,   // [64][64]
                       const float* __restrict__ w_vec2,   // [16][64]
                       const float* __restrict__ w_sca,    // [16][320]
                       const float* __restrict__ w_gate,   // [16][16]
                       _Float16*    __restrict__ Bh,
                       float*       __restrict__ fws)      // Bnorm | C2 | gateB
{
    __shared__ float Ash[HID_VEC];
    int tid = threadIdx.x;
    if (tid < HID_VEC) {
        float a = 0.f;
        #pragma unroll 8
        for (int c = 0; c < HID_VEC; ++c) a += w_vec1[tid*HID_VEC + c] * w_edge[c];
        Ash[tid] = a;
    }
    __syncthreads();
    if (tid < NUM_HEADS) {
        float bn = 0.f, cv = 0.f;
        #pragma unroll 8
        for (int h = 0; h < HID_VEC; ++h) {
            bn += w_sca[tid*320 + h] * fabsf(Ash[h]);
            cv += w_vec2[tid*HID_VEC + h] * Ash[h];
        }
        fws[tid]      = bn;        // Bnorm
        fws[16 + tid] = cv * cv;   // C^2
    }
    // Pack f16 B matrix: B[k][n] = w_sca[n][64+k], WMMA 16-bit B layout
    for (int idx = tid; idx < 8*32*16; idx += 256) {
        int kc   = idx >> 9;
        int lane = (idx >> 4) & 31;
        int t    = idx & 15;
        int g    = lane >> 4;
        int n    = lane & 15;
        int k    = kc*32 + (t < 8 ? t : t + 8) + 8*g;   // 0..255
        Bh[idx] = (_Float16)w_sca[n*320 + 64 + k];
    }
    // Pack f32 gate B: B[k][n] = w_gate[n][k], 16x16x4 layout, 4 chunks
    if (tid < 256) {
        int q    = tid >> 6;
        int lane = (tid >> 1) & 31;
        int j    = tid & 1;
        int n    = lane & 15;
        int k    = q*4 + 2*(lane >> 4) + j;
        fws[32 + tid] = w_gate[n*16 + k];
    }
}

__global__ __launch_bounds__(256)
void edge_bias_kernel(const int*   __restrict__ eidx,   // [2][E]
                      const float* __restrict__ feat,   // [E][5]
                      const float* __restrict__ pos,    // [N][3]
                      const v16h*  __restrict__ Bh,     // [8][32] v16h
                      const float* __restrict__ fws,    // Bnorm|C2|gateB
                      const float* __restrict__ b_gate, // [16]
                      float*       __restrict__ out_sca,
                      float*       __restrict__ out_vec,
                      int E)
{
    __shared__ float s_sm[WAVES_PER_BLOCK][TILE_E];
    __shared__ float sca_sm[WAVES_PER_BLOCK][TILE_E][17];   // padded: 17 coprime to 64 banks

    const int wave = threadIdx.x >> 5;
    const int lane = threadIdx.x & 31;
    const int g    = lane >> 4;       // half-wave
    const int m0   = lane & 15;       // edge-in-tile (phase 1) / head (phase 2)
    const int base = (blockIdx.x * WAVES_PER_BLOCK + wave) * TILE_E;

    // ---- phase 1: per-edge distance + s ----
    int e  = base + m0;
    int ec = (e < E) ? e : (E - 1);
    int na = eidx[ec];
    int nb = eidx[E + ec];
    float dx = pos[na*3+0] - pos[nb*3+0];
    float dy = pos[na*3+1] - pos[nb*3+1];
    float dz = pos[na*3+2] - pos[nb*3+2];
    float d  = sqrtf(dx*dx + dy*dy + dz*dz);
    float s  = d / (d + 1e-7f);
    if (g == 0) s_sm[wave][m0] = s;

    float f5[5] = {0.f, 0.f, 0.f, 0.f, 0.f};
    if (g == 1) {
        const float* fp = feat + ec*5;
        f5[0]=fp[0]; f5[1]=fp[1]; f5[2]=fp[2]; f5[3]=fp[3]; f5[4]=fp[4];
    }
    __syncthreads();

    // ---- phase 2: [16e x 256] @ [256 x 16h] via 8x wmma f16 ----
    const float step  = 10.0f / 250.0f;
    const float coeff = -0.5f / (step * step);
    v8f acc = {};
    #pragma unroll
    for (int kc = 0; kc < 8; ++kc) {
        v16h a;
        #pragma unroll
        for (int t = 0; t < 16; ++t) {
            int k = kc*32 + (t < 8 ? t : t + 8) + 8*g;   // feature index 0..255
            float x  = d - (float)k * step;
            float ga = __expf(coeff * x * x);
            float val = ga;
            if (kc == 7 && t >= 11) {                    // k>=251 only when g==1
                val = g ? f5[t - 11] : ga;
            }
            a[t] = (_Float16)val;
        }
        v16h b = Bh[kc*32 + lane];
        acc = __builtin_amdgcn_wmma_f32_16x16x32_f16(
                  false, a, false, b, (short)0, acc, false, false);
    }

    // ---- phase 3: out_sca = acc + Bnorm[o]*s[m]  (D: head=lane&15, edge=r+8g)
    const int   o  = m0;
    const float Bn = fws[o];
    #pragma unroll
    for (int r = 0; r < 8; ++r) {
        int m = r + 8*g;
        sca_sm[wave][m][o] = acc[r] + Bn * s_sm[wave][m];
    }
    __syncthreads();

    // ---- phase 4: vectorized out_sca store (lane: edge=m0, heads h0..h0+7)
    {
        int eo = base + m0;
        if (eo < E) {
            int h0 = g * 8;
            float4 v0 = make_float4(sca_sm[wave][m0][h0+0], sca_sm[wave][m0][h0+1],
                                    sca_sm[wave][m0][h0+2], sca_sm[wave][m0][h0+3]);
            float4 v1 = make_float4(sca_sm[wave][m0][h0+4], sca_sm[wave][m0][h0+5],
                                    sca_sm[wave][m0][h0+6], sca_sm[wave][m0][h0+7]);
            *(float4*)(out_sca + (size_t)eo*16 + h0)     = v0;
            *(float4*)(out_sca + (size_t)eo*16 + h0 + 4) = v1;
        }
    }

    // ---- phase 5: gates via exact f32 WMMA: z = out_sca @ w_gate^T + b_gate
    v8f acc2;
    {
        float bg = b_gate[o];
        #pragma unroll
        for (int r = 0; r < 8; ++r) acc2[r] = bg;
    }
    const float2* gB = (const float2*)(fws + 32);
    #pragma unroll
    for (int q = 0; q < 4; ++q) {
        int kb = q*4 + 2*g;
        v2f va, vb;
        va[0] = sca_sm[wave][m0][kb];
        va[1] = sca_sm[wave][m0][kb + 1];
        float2 bw = gB[q*32 + lane];
        vb[0] = bw.x; vb[1] = bw.y;
        acc2 = __builtin_amdgcn_wmma_f32_16x16x4_f32(
                   false, va, false, vb, (short)0, acc2, false, false);
    }

    // ---- phase 6: output_vec[e,o] = sigmoid(z)^2 * C2[o] * s^2 ----
    const float c2o = fws[16 + o];
    #pragma unroll
    for (int r = 0; r < 8; ++r) {
        int m  = r + 8*g;
        int eo = base + m;
        if (eo < E) {
            float z    = acc2[r];
            float gate = 1.0f / (1.0f + __expf(-z));
            float sm   = s_sm[wave][m];
            out_vec[(size_t)eo*16 + o] = gate * gate * c2o * sm * sm;
        }
    }
}

extern "C" void kernel_launch(void* const* d_in, const int* in_sizes, int n_in,
                              void* d_out, int out_size, void* d_ws, size_t ws_size,
                              hipStream_t stream) {
    const int*   eidx   = (const int*)  d_in[0];   // [2][E]
    const float* feat   = (const float*)d_in[1];   // [E][5]
    const float* pos    = (const float*)d_in[2];   // [N][3]
    const float* w_edge = (const float*)d_in[3];   // [64][1]
    const float* w_vec1 = (const float*)d_in[4];   // [64][64]
    const float* w_vec2 = (const float*)d_in[5];   // [16][64]
    const float* w_sca  = (const float*)d_in[6];   // [16][320]
    const float* w_gate = (const float*)d_in[7];   // [16][16]
    const float* b_gate = (const float*)d_in[8];   // [16]

    const int E = in_sizes[0] / 2;

    _Float16* Bh  = (_Float16*)d_ws;
    float*    fws = (float*)((char*)d_ws + 8192);

    float* out_sca = (float*)d_out;            // [E][16]
    float* out_vec = out_sca + (size_t)E * 16; // [E][16]

    precompute_kernel<<<1, 256, 0, stream>>>(w_edge, w_vec1, w_vec2, w_sca, w_gate,
                                             Bh, fws);

    int edges_per_block = WAVES_PER_BLOCK * TILE_E; // 128
    int nblk = (E + edges_per_block - 1) / edges_per_block;
    edge_bias_kernel<<<nblk, 256, 0, stream>>>(eidx, feat, pos, (const v16h*)Bh,
                                               fws, b_gate, out_sca, out_vec, E);
}
